// WaveRNN_6640019440484
// MI455X (gfx1250) — matrix operand order repaired and checked by reading, not compiled, not run on hardware
//
#include <hip/hip_runtime.h>
#include <hip/hip_bf16.h>
#include <math.h>

// ---------------------------------------------------------------------------
// WaveRNN sampling for MI455X (gfx1250): persistent wave32 kernel, fp16 WMMA.
//   hidden GEMM  : 168 Mtiles x 2 Nhalves = 336 wave jobs = 84 WGs x waves 0-3
//   waves 4-7    : cooperative 128B-line prefetch of conditional (t%32==0)
//   serial tail  : GRU + FC1 + FC2 + argmax (coarse, then fine) in WG0 using
//                  all 8 waves; activations packed fp16 in LDS
//   sync         : only 2 grid barriers per timestep
//   weights      : pre-packed to CDNA5 WMMA A/B lane layouts (fp16, ~6MB,
//                  L2-resident across the whole run)
// ---------------------------------------------------------------------------

typedef _Float16 f16;
typedef __attribute__((ext_vector_type(16))) _Float16 v16h;
typedef __attribute__((ext_vector_type(8)))  float    v8f;

#define SIZE_  896
#define HALF_  448
#define BINS_  256
#define BATCH_ 32
#define TLEN_  1024
#define NWG_   84     // 84 WGs x 4 GEMM waves = 336 == 168 Mtiles * 2 Nhalves
#define NTHR_  256    // 8 waves per WG (wave32)

// ---- workspace byte offsets (all 256B aligned) ----------------------------
#define WS_BAR      0u          // 2 uints: grid barrier {count, generation}
#define WS_HIDPACK  256u        // 896x32 f16, WMMA-B tile layout (57344 B)
#define WS_HP       57600u      // 2688x32 f32 hp matrix (344064 B)
#define WS_BHH      401664u     // 2688 f32 reordered b_hh
#define WS_CHID     412416u     // 448x32 f32 coarse hidden
#define WS_FHID     469760u     // 448x32 f32 fine hidden
#define WS_ZERO_END 527104u     // memset 0 .. here each launch
#define WS_WHHP     527104u     // 2688x896 f16 packed W_hh_r  (4816896 B)
#define WS_WC1P     5344000u    // 448x448 f16 packed wc1
#define WS_WF1P     5745408u    // 448x448 f16 packed wf1
#define WS_WC2P     6146816u    // 256x448 f16 packed wc2
#define WS_WF2P     6376192u    // 256x448 f16 packed wf2
#define WS_TOTAL    6605568u

static_assert(NWG_ * 4 == (SIZE_ * 3 / 16) * 2, "phase-1 job mapping");

// ---------------------------------------------------------------------------
// CDNA5 WMMA 16-bit B-matrix (32x16, KxN) lane layout:
//   lane L holds N = L%16; with g = L/16, half h of a lane is
//   K = h<8 ? 8g + h : 16 + 8g + (h-8)   (matches the A-tile K order)
// packed tile = 512 halves, tiles indexed (ktile*2 + nhalf).
__device__ __forceinline__ int packBidx(int K, int N) {
  int kt = K >> 5, kr = K & 31, nh = N >> 4, Nl = N & 15;
  int lane = (((kr >> 3) & 1) << 4) | Nl;
  int h = (kr & 7) | ((kr >> 4) << 3);
  return ((kt * 2 + nh) << 9) + (lane << 4) + h;
}

// ---------------------------------------------------------------------------
// Pack a row-major fp32 matrix (Mdim x Kdim) into WMMA 16-bit A-tiles.
// A 16x32 f16 layout (CDNA5 ISA): lanes 0-15 are M, VGPR v (half pair p):
//   lanes 0-15 : K = v<4 ? 2v+p      : 16 + 2(v-4)+p
//   lanes 16-31: K = v<4 ? 8 + 2v+p  : 24 + 2(v-4)+p
__global__ __launch_bounds__(256) void pack_a_kernel(const float* __restrict__ src,
                                                     f16* __restrict__ dst,
                                                     int Mdim, int Kdim) {
  int idx = blockIdx.x * 256 + threadIdx.x;
  if (idx >= Mdim * Kdim) return;
  int KT = Kdim >> 5;
  int tile = idx >> 9, within = idx & 511;
  int lane = within >> 4, h = within & 15;
  int mt = tile / KT, kt = tile % KT;
  int Ml = lane & 15, hi = lane >> 4;
  int v = h >> 1, p = h & 1;
  int base = (v < 4) ? (hi * 8) : (16 + hi * 8);
  int K = kt * 32 + base + ((v & 3) << 1) + p;
  int M = mt * 16 + Ml;
  dst[idx] = (f16)src[M * Kdim + K];
}

// Same packing for w_hh but with the _reorder row permutation
// ([cr,fr,cu,fu,ce,fe] -> [cr,cu,ce,fr,fu,fe]); also reorders b_hh.
__global__ __launch_bounds__(256) void pack_whh_kernel(const float* __restrict__ whh,
                                                       const float* __restrict__ bhh,
                                                       f16* __restrict__ dst,
                                                       float* __restrict__ bdst) {
  int idx = blockIdx.x * 256 + threadIdx.x;
  const int chunkMap[6] = {0, 2, 4, 1, 3, 5};   // dest chunk -> src chunk
  if (idx < 3 * SIZE_) {
    bdst[idx] = bhh[chunkMap[idx / HALF_] * HALF_ + (idx % HALF_)];
  }
  if (idx >= 3 * SIZE_ * SIZE_) return;
  int tile = idx >> 9, within = idx & 511;
  int lane = within >> 4, h = within & 15;
  int mt = tile / 28, kt = tile % 28;
  int Ml = lane & 15, hi = lane >> 4;
  int v = h >> 1, p = h & 1;
  int base = (v < 4) ? (hi * 8) : (16 + hi * 8);
  int K = kt * 32 + base + ((v & 3) << 1) + p;
  int dm = mt * 16 + Ml;
  int srcRow = chunkMap[dm / HALF_] * HALF_ + (dm % HALF_);
  dst[idx] = (f16)whh[srcRow * SIZE_ + K];
}

// ---------------------------------------------------------------------------
__device__ __forceinline__ void grid_barrier(unsigned* cnt, unsigned* gen) {
  __threadfence();
  __syncthreads();
  if (threadIdx.x == 0) {
    unsigned g = __hip_atomic_load(gen, __ATOMIC_RELAXED, __HIP_MEMORY_SCOPE_AGENT);
    unsigned a = __hip_atomic_fetch_add(cnt, 1u, __ATOMIC_ACQ_REL, __HIP_MEMORY_SCOPE_AGENT);
    if (a == NWG_ - 1) {
      __hip_atomic_store(cnt, 0u, __ATOMIC_RELAXED, __HIP_MEMORY_SCOPE_AGENT);
      __hip_atomic_fetch_add(gen, 1u, __ATOMIC_RELEASE, __HIP_MEMORY_SCOPE_AGENT);
    } else {
      while (__hip_atomic_load(gen, __ATOMIC_ACQUIRE, __HIP_MEMORY_SCOPE_AGENT) == g)
        __builtin_amdgcn_s_sleep(1);
    }
  }
  __syncthreads();
}

// FC layer on one workgroup (8 waves): D = A(M x 448) * Bpack + bias.
// If reluPackOut != null: relu and repack fp16 B-tiles (for the next layer).
// Else: write f32 logits[M][32].
__device__ __forceinline__ void fc_layer(const f16* __restrict__ Apack,
                                         const f16* __restrict__ Bpack,
                                         const float* __restrict__ bias,
                                         int Mtiles, int Ktiles,
                                         f16* __restrict__ reluPackOut,
                                         float* __restrict__ logitsOut,
                                         int wave, int lane) {
  int Nl = lane & 15, hi = lane >> 4;
  for (int j = wave; j < Mtiles * 2; j += 8) {
    int mt = j >> 1, nh = j & 1;
    v8f acc = {};
    const f16* Ab = Apack + (size_t)mt * Ktiles * 512;
    for (int kt = 0; kt < Ktiles; ++kt) {
      v16h a = *(const v16h*)(Ab + kt * 512 + lane * 16);
      v16h b = *(const v16h*)(Bpack + (kt * 2 + nh) * 512 + lane * 16);
      acc = __builtin_amdgcn_wmma_f32_16x16x32_f16(false, a, false, b, (short)0, acc,
                                                   false, false);
    }
    int N = nh * 16 + Nl;
#pragma unroll
    for (int r = 0; r < 8; ++r) {
      int o = mt * 16 + r + hi * 8;   // C layout: lanes16-31 hold M = r+8
      float vv = acc[r] + bias[o];
      if (reluPackOut) {
        vv = vv > 0.f ? vv : 0.f;
        reluPackOut[packBidx(o, N)] = (f16)vv;
      } else {
        logitsOut[o * 32 + N] = vv;
      }
    }
  }
}

__device__ __forceinline__ float sigmoidf_(float x) { return 1.f / (1.f + __expf(-x)); }

// ---------------------------------------------------------------------------
__global__ __launch_bounds__(NTHR_) void wavernn_persistent(
    const float* __restrict__ cond,
    const float* __restrict__ wci, const float* __restrict__ wfi,
    const float* __restrict__ b_ih,
    const float* __restrict__ bc1, const float* __restrict__ bc2,
    const float* __restrict__ bf1, const float* __restrict__ bf2,
    float* __restrict__ out, char* __restrict__ ws) {
  // 32KB region shared by {packed GRU output (28KB f16), logits (32KB f32)}
  __shared__ __align__(32) char s_u0_raw[32768];
  __shared__ __align__(32) f16  s_h1[14336];      // 28KB packed FC1 output
  __shared__ int s_prevC[32], s_prevF[32], s_newC[32];
  f16*   hPack  = (f16*)s_u0_raw;
  float* logits = (float*)s_u0_raw;

  f16*      hidpack = (f16*)(ws + WS_HIDPACK);
  float*    hp      = (float*)(ws + WS_HP);
  float*    bhhr    = (float*)(ws + WS_BHH);
  float*    chid    = (float*)(ws + WS_CHID);
  float*    fhid    = (float*)(ws + WS_FHID);
  const f16* whhp   = (const f16*)(ws + WS_WHHP);
  const f16* wc1p   = (const f16*)(ws + WS_WC1P);
  const f16* wf1p   = (const f16*)(ws + WS_WF1P);
  const f16* wc2p   = (const f16*)(ws + WS_WC2P);
  const f16* wf2p   = (const f16*)(ws + WS_WF2P);
  unsigned* barCnt  = (unsigned*)(ws + WS_BAR);
  unsigned* barGen  = barCnt + 1;

  const int tid  = threadIdx.x;
  const int wave = tid >> 5, lane = tid & 31;
  const int Nl   = lane & 15, hi = lane >> 4;
  const int wg   = blockIdx.x;
  const float sc = 2.0f / 255.0f;

  if (wg == 0 && tid < 32) { s_prevC[tid] = BINS_ / 2; s_prevF[tid] = 0; }
  __syncthreads();

  for (int t = 0; t < TLEN_; ++t) {
    // ---- Phase 1: hp = W_hh_r @ [c_hid;f_hid] + b_hh_r -------------------
    if (wave < 4) {                     // 336 GEMM waves, one job each
      int job = wg * 4 + wave;
      int mt = job >> 1, nh = job & 1;
      const f16* Ab = whhp + (size_t)mt * 28 * 512;
      v8f acc = {};
      for (int kt = 0; kt < 28; ++kt) {
        v16h a = *(const v16h*)(Ab + kt * 512 + lane * 16);
        v16h b = *(const v16h*)(hidpack + (kt * 2 + nh) * 512 + lane * 16);
        acc = __builtin_amdgcn_wmma_f32_16x16x32_f16(false, a, false, b, (short)0, acc,
                                                     false, false);
      }
#pragma unroll
      for (int r = 0; r < 8; ++r) {
        int row = mt * 16 + r + hi * 8;
        hp[row * 32 + nh * 16 + Nl] = acc[r] + bhhr[row];
      }
    } else if ((t & 31) == 0) {
      // waves 4-7 of every WG: prefetch one 128B-line window of `conditional`
      // (each line holds 32 consecutive t values for one (b,row) pair, so this
      // covers steps t..t+31); ~8 global_prefetch_b8 per thread per window.
      int self = wg * 128 + (tid - 128);          // 0 .. 84*128-1
      for (int q = self; q < 2688 * 32; q += NWG_ * 128) {
        int b = q & 31, row = q >> 5;
        __builtin_prefetch(&cond[((size_t)b * 2688 + row) * 1024 + t], 0, 1);
      }
    }
    grid_barrier(barCnt, barGen);       // hp ready

    if (wg == 0) {
      // ================= coarse subnet =================
      for (int idx = tid; idx < HALF_ * 32; idx += NTHR_) {
        int i = idx >> 5, b = idx & 31;
        float scc = (float)s_prevC[b] * sc - 1.0f;
        float scf = (float)s_prevF[b] * sc - 1.0f;
        size_t cb = ((size_t)b * 2688) * 1024 + (size_t)t;
        float ipr = wci[i * 2] * scc + wci[i * 2 + 1] * scf +
                    cond[cb + (size_t)(i) * 1024] + b_ih[i];
        float ipu = wci[(448 + i) * 2] * scc + wci[(448 + i) * 2 + 1] * scf +
                    cond[cb + (size_t)(896 + i) * 1024] + b_ih[896 + i];
        float ipe = wci[(896 + i) * 2] * scc + wci[(896 + i) * 2 + 1] * scf +
                    cond[cb + (size_t)(1792 + i) * 1024] + b_ih[1792 + i];
        float r = sigmoidf_(hp[i * 32 + b] + ipr);
        float u = sigmoidf_(hp[(448 + i) * 32 + b] + ipu);
        float e = tanhf(r * hp[(896 + i) * 32 + b] + ipe);
        float hn = u * chid[idx] + (1.f - u) * e;
        chid[idx] = hn;
        f16 hh = (f16)hn;
        hPack[packBidx(i, b)] = hh;       // LDS  : FC1 input
        hidpack[packBidx(i, b)] = hh;     // global: next-step GEMM (K = i)
      }
      __syncthreads();
      fc_layer(wc1p, hPack, bc1, 28, 14, s_h1, nullptr, wave, lane);
      __syncthreads();
      fc_layer(wc2p, s_h1, bc2, 16, 14, nullptr, logits, wave, lane);
      __syncthreads();
      if (tid < 32) {
        int best = 0; float bv = logits[tid];
        for (int m = 1; m < BINS_; ++m) {
          float v = logits[m * 32 + tid];
          if (v > bv) { bv = v; best = m; }
        }
        s_newC[tid] = best;
        out[tid * TLEN_ + t] = (float)best;           // oc.T[b][t]
      }
      __syncthreads();
      // ================= fine subnet =================
      for (int idx = tid; idx < HALF_ * 32; idx += NTHR_) {
        int i = idx >> 5, b = idx & 31;
        float scc = (float)s_prevC[b] * sc - 1.0f;
        float scf = (float)s_prevF[b] * sc - 1.0f;
        float scn = (float)s_newC[b] * sc - 1.0f;
        size_t cb = ((size_t)b * 2688) * 1024 + (size_t)t;
        float ipr = wfi[i * 3] * scc + wfi[i * 3 + 1] * scf + wfi[i * 3 + 2] * scn +
                    cond[cb + (size_t)(448 + i) * 1024] + b_ih[448 + i];
        float ipu = wfi[(448 + i) * 3] * scc + wfi[(448 + i) * 3 + 1] * scf +
                    wfi[(448 + i) * 3 + 2] * scn +
                    cond[cb + (size_t)(1344 + i) * 1024] + b_ih[1344 + i];
        float ipe = wfi[(896 + i) * 3] * scc + wfi[(896 + i) * 3 + 1] * scf +
                    wfi[(896 + i) * 3 + 2] * scn +
                    cond[cb + (size_t)(2240 + i) * 1024] + b_ih[2240 + i];
        float r = sigmoidf_(hp[(1344 + i) * 32 + b] + ipr);
        float u = sigmoidf_(hp[(1792 + i) * 32 + b] + ipu);
        float e = tanhf(r * hp[(2240 + i) * 32 + b] + ipe);
        float hn = u * fhid[idx] + (1.f - u) * e;
        fhid[idx] = hn;
        f16 hh = (f16)hn;
        hPack[packBidx(i, b)] = hh;
        hidpack[packBidx(448 + i, b)] = hh;           // K = 448+i
      }
      __syncthreads();
      fc_layer(wf1p, hPack, bf1, 28, 14, s_h1, nullptr, wave, lane);
      __syncthreads();
      fc_layer(wf2p, s_h1, bf2, 16, 14, nullptr, logits, wave, lane);
      __syncthreads();
      if (tid < 32) {
        int best = 0; float bv = logits[tid];
        for (int m = 1; m < BINS_; ++m) {
          float v = logits[m * 32 + tid];
          if (v > bv) { bv = v; best = m; }
        }
        out[32768 + tid * TLEN_ + t] = (float)best;   // of.T[b][t]
        s_prevC[tid] = s_newC[tid];
        s_prevF[tid] = best;
      }
      __syncthreads();
    }
    grid_barrier(barCnt, barGen);       // hidden/samples ready for next step
  }

  if (wg == 0) {
    for (int idx = tid; idx < SIZE_ * 32; idx += NTHR_) {
      int i = idx >> 5, b = idx & 31;
      out[65536 + idx] = (i < HALF_) ? chid[i * 32 + b] : fhid[(i - HALF_) * 32 + b];
    }
  }
}

// ---------------------------------------------------------------------------
extern "C" void kernel_launch(void* const* d_in, const int* in_sizes, int n_in,
                              void* d_out, int out_size, void* d_ws, size_t ws_size,
                              hipStream_t stream) {
  const float* cond = (const float*)d_in[0];
  const float* wci  = (const float*)d_in[1];
  const float* wfi  = (const float*)d_in[2];
  const float* whh  = (const float*)d_in[3];
  const float* bih  = (const float*)d_in[4];
  const float* bhh  = (const float*)d_in[5];
  const float* wc1  = (const float*)d_in[6];
  const float* bc1  = (const float*)d_in[7];
  const float* wc2  = (const float*)d_in[8];
  const float* bc2  = (const float*)d_in[9];
  const float* wf1  = (const float*)d_in[10];
  const float* bf1  = (const float*)d_in[11];
  const float* wf2  = (const float*)d_in[12];
  const float* bf2  = (const float*)d_in[13];
  float* out = (float*)d_out;
  char*  ws  = (char*)d_ws;
  (void)in_sizes; (void)n_in; (void)out_size; (void)ws_size;

  // zero barrier state, packed hidden, GRU states each launch (graph-safe)
  hipMemsetAsync(ws, 0, WS_ZERO_END, stream);

  pack_whh_kernel<<<(3 * SIZE_ * SIZE_ + 255) / 256, 256, 0, stream>>>(
      whh, bhh, (f16*)(ws + WS_WHHP), (float*)(ws + WS_BHH));
  pack_a_kernel<<<(HALF_ * HALF_ + 255) / 256, 256, 0, stream>>>(
      wc1, (f16*)(ws + WS_WC1P), HALF_, HALF_);
  pack_a_kernel<<<(HALF_ * HALF_ + 255) / 256, 256, 0, stream>>>(
      wf1, (f16*)(ws + WS_WF1P), HALF_, HALF_);
  pack_a_kernel<<<(BINS_ * HALF_ + 255) / 256, 256, 0, stream>>>(
      wc2, (f16*)(ws + WS_WC2P), BINS_, HALF_);
  pack_a_kernel<<<(BINS_ * HALF_ + 255) / 256, 256, 0, stream>>>(
      wf2, (f16*)(ws + WS_WF2P), BINS_, HALF_);

  wavernn_persistent<<<NWG_, NTHR_, 0, stream>>>(
      cond, wci, wfi, bih, bc1, bc2, bf1, bf2, out, ws);
}
